// MaskedEdgeAttention_3582002725441
// MI455X (gfx1250) — matrix operand order repaired and checked by reading, not compiled, not run on hardware
//
#include <hip/hip_runtime.h>

#define SEQ    512
#define BATCH  64
#define DIM    512
#define MAXLEN 512

typedef __attribute__((ext_vector_type(16))) _Float16 v16h;
typedef __attribute__((ext_vector_type(8)))  float    v8f;

struct alignas(16) H8      { _Float16 h[8]; };
struct alignas(16) H16pair { H8 lo, hi; };

__device__ __forceinline__ v16h make_frag(H8 lo, H8 hi) {
    H16pair p{lo, hi};
    return __builtin_bit_cast(v16h, p);
}

__device__ __forceinline__ H8 cvt8(float4 u, float4 v) {
    H8 r;
    r.h[0] = (_Float16)u.x; r.h[1] = (_Float16)u.y;
    r.h[2] = (_Float16)u.z; r.h[3] = (_Float16)u.w;
    r.h[4] = (_Float16)v.x; r.h[5] = (_Float16)v.y;
    r.h[6] = (_Float16)v.z; r.h[7] = (_Float16)v.w;
    return r;
}

// per-lane 16B async copy global -> LDS, tracked by ASYNCcnt
#define CP16(ldsoff, gaddr)                                                    \
    asm volatile("global_load_async_to_lds_b128 %0, %1, off"                   \
                 :: "v"(ldsoff), "v"(gaddr) : "memory")

// ---------------------------------------------------------------------------
// One-shot f32 -> f16 conversion (enables raw async tile copies in the GEMM).
// M (s,b,d) f32  ->  M16 [b][s][d] f16 ;  W (m,d) f32 -> W16 [m][d] f16
// ---------------------------------------------------------------------------
__global__ __launch_bounds__(256)
void cvt_M(const float* __restrict__ M, _Float16* __restrict__ M16) {
    const size_t idx = (size_t)blockIdx.x * 256 + threadIdx.x;   // 8 elems/thread
    const int d = (int)(idx & 63) * 8;
    const int s = (int)((idx >> 6) & (SEQ - 1));
    const int b = (int)(idx >> 15);
    const float* src = M + ((size_t)s * BATCH + b) * DIM + d;
    const float4 f0 = *(const float4*)src;
    const float4 f1 = *(const float4*)(src + 4);
    *(H8*)(M16 + ((size_t)b * SEQ + s) * DIM + d) = cvt8(f0, f1);
}

__global__ __launch_bounds__(256)
void cvt_W(const float* __restrict__ W, _Float16* __restrict__ W16) {
    const size_t idx = (size_t)blockIdx.x * 256 + threadIdx.x;
    const int d = (int)(idx & 63) * 8;
    const int m = (int)(idx >> 6);
    const float* src = W + (size_t)m * DIM + d;
    const float4 f0 = *(const float4*)src;
    const float4 f1 = *(const float4*)(src + 4);
    *(H8*)(W16 + (size_t)m * DIM + d) = cvt8(f0, f1);
}

// ---------------------------------------------------------------------------
// GEMM: L[b][m][s] = sum_d W[m][d] * M[s][b][d]   (f16 WMMA, f32 accum)
// Block 128m x 128s, 8 waves; wave tile 64m x 32s (4x2 WMMA tiles).
// K staged 32-wide via double-buffered global_load_async_to_lds_b128
// (ASYNCcnt), LDS rows padded to 40 halves (80B) -> conflict-free 16B loads.
// ---------------------------------------------------------------------------
__global__ __launch_bounds__(256)
void gemm_logits(const _Float16* __restrict__ W16, const _Float16* __restrict__ M16,
                 float* __restrict__ L) {
    // [buf 0|1] x { A tile 5120 halves | B tile 5120 halves } = 40960 bytes
    __shared__ alignas(16) _Float16 Sh[2 * 10240];

    const int tid  = threadIdx.x;
    const int b    = blockIdx.z;
    const int m0   = blockIdx.y * 128;
    const int s0   = blockIdx.x * 128;
    const int lane = tid & 31;
    const int wave = tid >> 5;
    const int wm   = (wave & 1) * 64;     // wave m offset
    const int wso  = (wave >> 1) * 32;    // wave s offset
    const int r16  = lane & 15;
    const int hi   = lane >> 4;

    const _Float16* Msrc = M16 + (size_t)b * SEQ * DIM;

    // async-copy assignment: 512 16B-chunks per tile; chunk c = wave*64+{lane,lane+32}
    const int c0 = wave * 64 + lane;
    const int c1 = c0 + 32;
    const int rA0 = c0 >> 2, e0 = (c0 & 3) * 8;   // row, half-offset within row
    const int rA1 = c1 >> 2, e1 = (c1 & 3) * 8;

    const uint64_t gA0 = (uint64_t)(uintptr_t)(W16  + (size_t)(m0 + rA0) * DIM + e0);
    const uint64_t gA1 = (uint64_t)(uintptr_t)(W16  + (size_t)(m0 + rA1) * DIM + e1);
    const uint64_t gB0 = (uint64_t)(uintptr_t)(Msrc + (size_t)(s0 + rA0) * DIM + e0);
    const uint64_t gB1 = (uint64_t)(uintptr_t)(Msrc + (size_t)(s0 + rA1) * DIM + e1);

    const uint32_t ldsBase = (uint32_t)(uintptr_t)Sh;
    const uint32_t lA0 = ldsBase + (uint32_t)(rA0 * 80 + e0 * 2);
    const uint32_t lA1 = ldsBase + (uint32_t)(rA1 * 80 + e1 * 2);
    const uint32_t lB0 = lA0 + 10240;
    const uint32_t lB1 = lA1 + 10240;

    const v8f vzero = {0.f, 0.f, 0.f, 0.f, 0.f, 0.f, 0.f, 0.f};
    v8f acc[4][2];
#pragma unroll
    for (int i = 0; i < 4; ++i)
#pragma unroll
        for (int j = 0; j < 2; ++j) acc[i][j] = vzero;

    // prologue: stage K-slab 0 into buffer 0
    CP16(lA0, gA0); CP16(lA1, gA1); CP16(lB0, gB0); CP16(lB1, gB1);

    for (int it = 0; it < DIM / 32; ++it) {
        const int buf = it & 1;
        asm volatile("s_wait_asynccnt 0x0" ::: "memory");   // own copies for `buf` done
        __syncthreads();                                    // all waves' tiles resident;
                                                            // everyone done reading buf^1
        if (it < DIM / 32 - 1) {                            // prefetch next slab
            const uint64_t ko = (uint64_t)(it + 1) * 64;    // 32 halves * 2B
            const uint32_t bo = (uint32_t)((buf ^ 1) * 20480);
            CP16(lA0 + bo, gA0 + ko); CP16(lA1 + bo, gA1 + ko);
            CP16(lB0 + bo, gB0 + ko); CP16(lB1 + bo, gB1 + ko);
        }

        const _Float16* Sb = Sh + (size_t)buf * 10240;
        // A (16x32 f16): lanes<16 K={0..7,16..23}, lanes>=16 K={8..15,24..31}
        v16h af[4];
#pragma unroll
        for (int i = 0; i < 4; ++i) {
            const _Float16* p = &Sb[(wm + i * 16 + r16) * 40 + hi * 8];
            af[i] = make_frag(*(const H8*)p, *(const H8*)(p + 16));
        }
        // B (32x16 f16): lane holds column n=lane%16, lanes<16 K={0..7,16..23}, lanes>=16 K={8..15,24..31}
        v16h bf[2];
#pragma unroll
        for (int j = 0; j < 2; ++j) {
            const _Float16* p = &Sb[5120 + (wso + j * 16 + r16) * 40 + hi * 8];
            bf[j] = make_frag(*(const H8*)p, *(const H8*)(p + 16));
        }
#pragma unroll
        for (int i = 0; i < 4; ++i)
#pragma unroll
            for (int j = 0; j < 2; ++j)
                acc[i][j] = __builtin_amdgcn_wmma_f32_16x16x32_f16(
                    false, af[i], false, bf[j], (short)0, acc[i][j], false, false);
    }

    // epilogue: C/D layout: VGPR r -> (M = r + 8*hi, N = lane%16)
    float* Lb = L + (size_t)b * MAXLEN * SEQ;
#pragma unroll
    for (int i = 0; i < 4; ++i)
#pragma unroll
        for (int j = 0; j < 2; ++j) {
            const int gs = s0 + wso + j * 16 + r16;
#pragma unroll
            for (int r = 0; r < 8; ++r) {
                const int gm = m0 + wm + i * 16 + r + hi * 8;
                Lb[(size_t)gm * SEQ + gs] = acc[i][j][r];
            }
        }
}

// ---------------------------------------------------------------------------
// Softmax over s for each (b,m) row (one wave per row); also zero d_out.
// ---------------------------------------------------------------------------
__global__ __launch_bounds__(256)
void softmax_rows(float* __restrict__ L, float* __restrict__ out) {
    const int row  = blockIdx.x * 8 + (threadIdx.x >> 5);
    const int lane = threadIdx.x & 31;
    const size_t base = (size_t)row * SEQ + lane;
    float x[16];
#pragma unroll
    for (int t = 0; t < 16; ++t) x[t] = L[base + t * 32];
    float mx = x[0];
#pragma unroll
    for (int t = 1; t < 16; ++t) mx = fmaxf(mx, x[t]);
#pragma unroll
    for (int o = 16; o > 0; o >>= 1) mx = fmaxf(mx, __shfl_xor(mx, o, 32));
    float s = 0.f;
#pragma unroll
    for (int t = 0; t < 16; ++t) { x[t] = __expf(x[t] - mx); s += x[t]; }
#pragma unroll
    for (int o = 16; o > 0; o >>= 1) s += __shfl_xor(s, o, 32);
    const float inv = 1.0f / s;
#pragma unroll
    for (int t = 0; t < 16; ++t) {
        L[base + t * 32]   = x[t] * inv;   // alpha, in place
        out[base + t * 32] = 0.0f;         // default output (mask_copy == 0)
    }
}

// ---------------------------------------------------------------------------
// Edge mask handling (exact .set semantics, duplicate-safe)
// ---------------------------------------------------------------------------
__global__ void zero_mask(uint4* __restrict__ p) {
    p[(size_t)blockIdx.x * 256 + threadIdx.x] = make_uint4(0u, 0u, 0u, 0u);
}

__global__ void set_mask(const int* __restrict__ eb, const int* __restrict__ eu,
                         const int* __restrict__ ev, unsigned char* __restrict__ mask,
                         int E) {
    const int e = blockIdx.x * 256 + threadIdx.x;
    if (e < E)
        mask[((size_t)eb[e] * MAXLEN + eu[e]) * SEQ + ev[e]] = 1;
}

// sums[b,u] = sum_v (mask ? alpha : 1e-10f * alpha)   (one wave per row)
__global__ __launch_bounds__(256)
void row_sums(const float* __restrict__ L, const unsigned char* __restrict__ mask,
              float* __restrict__ sums) {
    const int row  = blockIdx.x * 8 + (threadIdx.x >> 5);
    const int lane = threadIdx.x & 31;
    const size_t base = (size_t)row * SEQ + lane;
    float acc = 0.f;
#pragma unroll
    for (int t = 0; t < 16; ++t) {
        const float a = L[base + t * 32];
        acc += mask[base + t * 32] ? a : 1e-10f * a;
    }
#pragma unroll
    for (int o = 16; o > 0; o >>= 1) acc += __shfl_xor(acc, o, 32);
    if (lane == 0) sums[row] = acc;
}

__global__ void scatter_edges(const int* __restrict__ eb, const int* __restrict__ eu,
                              const int* __restrict__ ev, const float* __restrict__ L,
                              const float* __restrict__ sums, float* __restrict__ out,
                              int E) {
    const int e = blockIdx.x * 256 + threadIdx.x;
    if (e < E) {
        const size_t row = (size_t)eb[e] * MAXLEN + eu[e];
        const size_t idx = row * SEQ + ev[e];
        out[idx] = L[idx] / sums[row];   // duplicates write identical values
    }
}

// ---------------------------------------------------------------------------
extern "C" void kernel_launch(void* const* d_in, const int* in_sizes, int n_in,
                              void* d_out, int out_size, void* d_ws, size_t ws_size,
                              hipStream_t stream) {
    const float* M  = (const float*)d_in[0];
    const float* W  = (const float*)d_in[1];
    // d_in[2] = lengths: unused by the reference
    const int*   eb = (const int*)d_in[3];
    const int*   eu = (const int*)d_in[4];
    const int*   ev = (const int*)d_in[5];
    const int    E  = in_sizes[3];
    float* out = (float*)d_out;

    char* ws = (char*)d_ws;
    float*         L    = (float*)ws;                               // 64 MB logits/alpha
    unsigned char* mask = (unsigned char*)(ws + 67108864);          // 16 MB byte mask
    float*         sums = (float*)(ws + 83886080);                  // 128 KB row sums
    _Float16*      M16  = (_Float16*)(ws + 84017152);               // 32 MB f16 M [b][s][d]
    _Float16*      W16  = (_Float16*)(ws + 117571584);              // 512 KB f16 W [m][d]

    cvt_M<<<(BATCH * SEQ * DIM) / 8 / 256, 256, 0, stream>>>(M, M16);
    cvt_W<<<(MAXLEN * DIM) / 8 / 256, 256, 0, stream>>>(W, W16);

    const int maskVec4 = (BATCH * MAXLEN * SEQ) / 16;
    zero_mask<<<maskVec4 / 256, 256, 0, stream>>>((uint4*)mask);
    set_mask<<<(E + 255) / 256, 256, 0, stream>>>(eb, eu, ev, mask, E);

    gemm_logits<<<dim3(SEQ / 128, MAXLEN / 128, BATCH), 256, 0, stream>>>(W16, M16, L);

    const int rows = BATCH * MAXLEN;
    softmax_rows<<<rows / 8, 256, 0, stream>>>(L, out);
    row_sums<<<rows / 8, 256, 0, stream>>>(L, mask, sums);
    scatter_edges<<<(E + 255) / 256, 256, 0, stream>>>(eb, eu, ev, L, sums, out, E);
}
